// MemoryEfficientLinearAttention_43069932045062
// MI455X (gfx1250) — compile-verified, matrix operand order checked by reading, not run on hardware
//
#include <hip/hip_runtime.h>
#include <stdint.h>

// ---------------------------------------------------------------------------
// MemoryEfficientLinearAttention for MI455X (gfx1250), wave32 + WMMA bf16.
// Pipeline: f32->bf16 cvt, 3x QKV GEMM (wmma bf16, double-buffered async-LDS),
//           exp feature map, per-head kv reduction (+k_sum via ones-fragment),
//           attention output (+denominator via ksum-fragment), O GEMM -> fp32.
// ---------------------------------------------------------------------------

#define BATCH 4
#define SEQ   8192
#define EMB   1024
#define HEADS 16
#define HDIM  64
#define MTOT  (BATCH * SEQ)   // 32768

typedef __bf16 v16bf __attribute__((ext_vector_type(16)));
typedef float  v8f   __attribute__((ext_vector_type(8)));

union Frag16 {
    v16bf v;
    uint4 q[2];
    unsigned short u[16];
};

// ---- CDNA5 async global->LDS path (guarded; falls back to sync copies) ----
#if defined(__AMDGCN__) && defined(__has_builtin)
#  if __has_builtin(__builtin_amdgcn_global_load_async_to_lds_b128) && \
      __has_builtin(__builtin_amdgcn_s_wait_asynccnt)
#    define USE_ASYNC_LDS 1
#  else
#    define USE_ASYNC_LDS 0
#  endif
#else
#  define USE_ASYNC_LDS 0
#endif

#if USE_ASYNC_LDS
typedef int async_v4i __attribute__((ext_vector_type(4)));
typedef __attribute__((address_space(1))) async_v4i* gptr_v4i;  // global int4*
typedef __attribute__((address_space(3))) async_v4i* lptr_v4i;  // LDS int4*
#endif

__device__ __forceinline__ void copy16_g2l(const unsigned short* g,
                                           unsigned short* l) {
#if USE_ASYNC_LDS
    // Generic LDS pointer -> AS(3): the LDS address is the low 32 bits of the generic address.
    __builtin_amdgcn_global_load_async_to_lds_b128(
        (gptr_v4i)(uintptr_t)g,
        (lptr_v4i)(uintptr_t)(uint32_t)(uintptr_t)l,
        0, 0);
#else
    *(uint4*)l = *(const uint4*)g;
#endif
}

__device__ __forceinline__ void wait_async_done() {
#if USE_ASYNC_LDS
    __builtin_amdgcn_s_wait_asynccnt(0);
#endif
}

__device__ __forceinline__ unsigned short f32_to_bf16(float f) {
    union { float f; uint32_t u; } c; c.f = f;
    uint32_t u = c.u;
    uint32_t r = (u + 0x7FFFu + ((u >> 16) & 1u)) >> 16;   // round-nearest-even
    return (unsigned short)r;
}
__device__ __forceinline__ float bf16_to_f32(unsigned short h) {
    union { uint32_t u; float f; } c; c.u = ((uint32_t)h) << 16;
    return c.f;
}

// ---------------------------------------------------------------------------
// 1) fp32 -> bf16 conversion (grid-stride)
// ---------------------------------------------------------------------------
__global__ void cvt_f32_bf16_kernel(const float* __restrict__ in,
                                    unsigned short* __restrict__ out, int n) {
    int i = blockIdx.x * blockDim.x + threadIdx.x;
    int stride = gridDim.x * blockDim.x;
    for (; i < n; i += stride) out[i] = f32_to_bf16(in[i]);
}

// ---------------------------------------------------------------------------
// 2) Generic GEMM: C(MxN) = A(MxK,bf16) * W(NxK,bf16)^T + bias(fp32)
//    128x128 tile, 256 threads = 8 waves, wave = 32x64 sub-tile (2x4 frags).
//    KSTEP=64, double-buffered LDS filled by async global->LDS loads.
//    OUT_BF16: 1 -> bf16 output, 0 -> fp32 output.
// ---------------------------------------------------------------------------
template <int OUT_BF16>
__global__ __launch_bounds__(256)
void gemm_bias_kernel(const unsigned short* __restrict__ A,
                      const unsigned short* __restrict__ W,
                      const float* __restrict__ bias,
                      void* __restrict__ Cout,
                      int M, int N, int K) {
    constexpr int KSTEP = 64;
    constexpr int LDA = 72;  // halfs per LDS row (144B, 16B-aligned, bank-spread)
    __shared__ __attribute__((aligned(16))) unsigned short As[2][128 * LDA];
    __shared__ __attribute__((aligned(16))) unsigned short Bs[2][128 * LDA];

    const int tid  = threadIdx.x;
    const int lane = tid & 31;
    const int wave = tid >> 5;
    const int wm   = wave & 3;   // 0..3 -> 32-row slab
    const int wn   = wave >> 2;  // 0..1 -> 64-col slab
    const int l15  = lane & 15;
    const int kb   = (lane >> 4) * 8;  // wave32 16-bit A/B fragment K-base
    const int tileN = blockIdx.x * 128;
    const int tileM = blockIdx.y * 128;

    // Stage one 128x64 bf16 tile of A and W into buffer `buf`.
    // 1024 16B-chunks per matrix; 4 per thread per matrix.
    auto issue_tile = [&](int k0, int buf) {
#pragma unroll
        for (int c = 0; c < 4; ++c) {
            int chunk = tid + c * 256;        // 0..1023
            int row = chunk >> 3;             // 0..127
            int sub = chunk & 7;              // 0..7 (8 halfs each)
            copy16_g2l(A + (size_t)(tileM + row) * K + k0 + sub * 8,
                       &As[buf][row * LDA + sub * 8]);
            copy16_g2l(W + (size_t)(tileN + row) * K + k0 + sub * 8,
                       &Bs[buf][row * LDA + sub * 8]);
        }
    };

    v8f acc[2][4];
#pragma unroll
    for (int mf = 0; mf < 2; ++mf)
#pragma unroll
        for (int nf = 0; nf < 4; ++nf)
            acc[mf][nf] = (v8f){0.f, 0.f, 0.f, 0.f, 0.f, 0.f, 0.f, 0.f};

    const int nIter = K / KSTEP;  // 16
    issue_tile(0, 0);

    for (int i = 0; i < nIter; ++i) {
        const int cur = i & 1;
        wait_async_done();   // my chunks of tile i are in LDS
        __syncthreads();     // everyone's chunks in; everyone done reading buf^1
        if (i + 1 < nIter) issue_tile((i + 1) * KSTEP, 1 - cur);  // overlaps math

#pragma unroll
        for (int kh = 0; kh < 2; ++kh) {  // two 32-wide K slices of the tile
            const int ko = kh * 32 + kb;
            Frag16 af[2], bfr[4];
#pragma unroll
            for (int mf = 0; mf < 2; ++mf) {
                int r = wm * 32 + mf * 16 + l15;
                af[mf].q[0] = *(const uint4*)(&As[cur][r * LDA + ko]);
                af[mf].q[1] = *(const uint4*)(&As[cur][r * LDA + 16 + ko]);
            }
#pragma unroll
            for (int nf = 0; nf < 4; ++nf) {
                int r = wn * 64 + nf * 16 + l15;
                bfr[nf].q[0] = *(const uint4*)(&Bs[cur][r * LDA + ko]);
                bfr[nf].q[1] = *(const uint4*)(&Bs[cur][r * LDA + 16 + ko]);
            }
#pragma unroll
            for (int mf = 0; mf < 2; ++mf)
#pragma unroll
                for (int nf = 0; nf < 4; ++nf)
                    acc[mf][nf] = __builtin_amdgcn_wmma_f32_16x16x32_bf16(
                        false, af[mf].v, false, bfr[nf].v, (short)0,
                        acc[mf][nf], false, false);
        }
    }

    // Epilogue: +bias, store (C layout: VGPR v -> M = v + 8*(lane>=16), N = l15)
#pragma unroll
    for (int nf = 0; nf < 4; ++nf) {
        int col = tileN + wn * 64 + nf * 16 + l15;
        float bv = bias[col];
#pragma unroll
        for (int mf = 0; mf < 2; ++mf) {
#pragma unroll
            for (int v = 0; v < 8; ++v) {
                int row = tileM + wm * 32 + mf * 16 + v + ((lane >> 4) << 3);
                float val = acc[mf][nf][v] + bv;
                if (OUT_BF16)
                    ((unsigned short*)Cout)[(size_t)row * N + col] = f32_to_bf16(val);
                else
                    ((float*)Cout)[(size_t)row * N + col] = val;
            }
        }
    }
}

// ---------------------------------------------------------------------------
// 3) Feature map: per 64-wide head segment, x <- exp(x - max(x)), in place.
//    One wave per segment; lane handles elements lane and lane+32.
// ---------------------------------------------------------------------------
__global__ __launch_bounds__(256)
void feature_exp_kernel(unsigned short* __restrict__ qk) {
    int wseg = (blockIdx.x * blockDim.x + threadIdx.x) >> 5;  // segment id
    int lane = threadIdx.x & 31;
    size_t base = (size_t)wseg * 64;
    float a = bf16_to_f32(qk[base + lane]);
    float b = bf16_to_f32(qk[base + lane + 32]);
    float m = fmaxf(a, b);
#pragma unroll
    for (int d = 16; d >= 1; d >>= 1) m = fmaxf(m, __shfl_xor(m, d, 32));
    qk[base + lane]      = f32_to_bf16(__expf(a - m));
    qk[base + lane + 32] = f32_to_bf16(__expf(b - m));
}

// ---------------------------------------------------------------------------
// 4) kv[b,h,d,e] = sum_s k_kern[s,d] * v[s,e]  (+ k_sum via ones fragment)
//    1 block per (b,h); 8 waves: (wave&3)->d-slab, (wave>>2)->s half; split-K
//    partials reduced through LDS. Tiles staged with async global->LDS.
// ---------------------------------------------------------------------------
__global__ __launch_bounds__(256)
void kv_reduce_kernel(const unsigned short* __restrict__ kk,
                      const unsigned short* __restrict__ vv,
                      unsigned short* __restrict__ kvout,   // (64,64,64) bf16
                      unsigned short* __restrict__ ksum) {  // (64,64)    bf16
    constexpr int LDT = 72;  // halfs per LDS row (144B, 16B-aligned)
    __shared__ __attribute__((aligned(16))) unsigned short kt[64 * LDT];
    __shared__ __attribute__((aligned(16))) unsigned short vt[64 * LDT];
    __shared__ float red[4][5][8][32];

    const int bh = blockIdx.x;
    const int b = bh >> 4, h = bh & 15;
    const int tid = threadIdx.x, lane = tid & 31, wave = tid >> 5;
    const int sgrp = wave >> 2;           // 0/1 -> s rows 0-31 / 32-63 of tile
    const int d0 = (wave & 3) * 16;
    const int l15 = lane & 15, kb = (lane >> 4) * 8;
    const int soff = sgrp * 32;

    v8f acc[4], accden;
#pragma unroll
    for (int nf = 0; nf < 4; ++nf)
        acc[nf] = (v8f){0.f, 0.f, 0.f, 0.f, 0.f, 0.f, 0.f, 0.f};
    accden = (v8f){0.f, 0.f, 0.f, 0.f, 0.f, 0.f, 0.f, 0.f};

    Frag16 bones;  // B ones fragment: column n=0 all 1.0 -> k_sum column
#pragma unroll
    for (int j = 0; j < 16; ++j) bones.u[j] = (l15 == 0) ? 0x3F80 : 0;

    for (int s0 = 0; s0 < SEQ; s0 += 64) {
        __syncthreads();
#pragma unroll
        for (int c = 0; c < 2; ++c) {
            int chunk = tid + c * 256;  // 0..511
            int row = chunk >> 3;       // 0..63
            int sub = chunk & 7;        // 0..7
            size_t g = (size_t)(b * SEQ + s0 + row) * EMB + h * 64 + sub * 8;
            copy16_g2l(kk + g, kt + row * LDT + sub * 8);
            copy16_g2l(vv + g, vt + row * LDT + sub * 8);
        }
        wait_async_done();
        __syncthreads();

        Frag16 af, bfr[4];
#pragma unroll
        for (int j = 0; j < 8; ++j) {  // A = k^T: A(d, s) = kt[s][d]
            af.u[j]     = kt[(soff + kb + j) * LDT + d0 + l15];
            af.u[8 + j] = kt[(soff + 16 + kb + j) * LDT + d0 + l15];
        }
#pragma unroll
        for (int nf = 0; nf < 4; ++nf)
#pragma unroll
            for (int j = 0; j < 8; ++j) {  // B = v: B(s, e) = vt[s][e]
                bfr[nf].u[j]     = vt[(soff + kb + j) * LDT + nf * 16 + l15];
                bfr[nf].u[8 + j] = vt[(soff + 16 + kb + j) * LDT + nf * 16 + l15];
            }
#pragma unroll
        for (int nf = 0; nf < 4; ++nf)
            acc[nf] = __builtin_amdgcn_wmma_f32_16x16x32_bf16(
                false, af.v, false, bfr[nf].v, (short)0, acc[nf], false, false);
        accden = __builtin_amdgcn_wmma_f32_16x16x32_bf16(
            false, af.v, false, bones.v, (short)0, accden, false, false);
    }

    __syncthreads();
    if (sgrp == 1) {
        int w = wave & 3;
#pragma unroll
        for (int nf = 0; nf < 4; ++nf)
#pragma unroll
            for (int v = 0; v < 8; ++v) red[w][nf][v][lane] = acc[nf][v];
#pragma unroll
        for (int v = 0; v < 8; ++v) red[w][4][v][lane] = accden[v];
    }
    __syncthreads();
    if (sgrp == 0) {
        int w = wave;
#pragma unroll
        for (int nf = 0; nf < 4; ++nf) {
#pragma unroll
            for (int v = 0; v < 8; ++v) {
                float val = acc[nf][v] + red[w][nf][v][lane];
                int d = d0 + v + ((lane >> 4) << 3);
                int e = nf * 16 + l15;
                kvout[(size_t)bh * 4096 + d * 64 + e] = f32_to_bf16(val);
            }
        }
        if (l15 == 0) {
#pragma unroll
            for (int v = 0; v < 8; ++v) {
                float val = accden[v] + red[w][4][v][lane];
                int d = d0 + v + ((lane >> 4) << 3);
                ksum[bh * 64 + d] = f32_to_bf16(val);
            }
        }
    }
}

// ---------------------------------------------------------------------------
// 5) y = (q_kern @ kv) / (q_kern . ksum + 1e-6); 32 blocks per (b,h),
//    wave = 32 q-rows; denominator via ksum B-fragment (column 0).
// ---------------------------------------------------------------------------
__global__ __launch_bounds__(256)
void attn_out_kernel(const unsigned short* __restrict__ qk,
                     const unsigned short* __restrict__ kv,
                     const unsigned short* __restrict__ ksum,
                     unsigned short* __restrict__ y) {
    constexpr int LDT = 72;
    __shared__ __attribute__((aligned(16))) unsigned short kvt[64 * LDT];
    __shared__ unsigned short kst[64];

    const int bid = blockIdx.x;
    const int bh = bid >> 5;   // 32 blocks per (b,h)
    const int tb = bid & 31;
    const int b = bh >> 4, h = bh & 15;
    const int tid = threadIdx.x, lane = tid & 31, wave = tid >> 5;
    const int l15 = lane & 15, kb = (lane >> 4) * 8;

#pragma unroll
    for (int c = 0; c < 2; ++c) {
        int chunk = tid + c * 256;  // 0..511
        int row = chunk >> 3, sub = chunk & 7;
        copy16_g2l(kv + (size_t)bh * 4096 + row * 64 + sub * 8,
                   kvt + row * LDT + sub * 8);
    }
    if (tid < 64) kst[tid] = ksum[bh * 64 + tid];
    wait_async_done();
    __syncthreads();

    const int srow0 = tb * 256 + wave * 32;
    v8f acc[2][4], accden[2];
#pragma unroll
    for (int mf = 0; mf < 2; ++mf) {
#pragma unroll
        for (int nf = 0; nf < 4; ++nf)
            acc[mf][nf] = (v8f){0.f, 0.f, 0.f, 0.f, 0.f, 0.f, 0.f, 0.f};
        accden[mf] = (v8f){0.f, 0.f, 0.f, 0.f, 0.f, 0.f, 0.f, 0.f};
    }

#pragma unroll
    for (int ks = 0; ks < 2; ++ks) {
        Frag16 bfr[4], bden;
#pragma unroll
        for (int nf = 0; nf < 4; ++nf)
#pragma unroll
            for (int j = 0; j < 8; ++j) {  // B(d, e) = kv[d][e]
                bfr[nf].u[j]     = kvt[(ks * 32 + kb + j) * LDT + nf * 16 + l15];
                bfr[nf].u[8 + j] = kvt[(ks * 32 + 16 + kb + j) * LDT + nf * 16 + l15];
            }
#pragma unroll
        for (int j = 0; j < 8; ++j) {  // denominator column: B(d,0)=ksum[d]
            bden.u[j]     = (l15 == 0) ? kst[ks * 32 + kb + j] : (unsigned short)0;
            bden.u[8 + j] = (l15 == 0) ? kst[ks * 32 + 16 + kb + j] : (unsigned short)0;
        }
#pragma unroll
        for (int mf = 0; mf < 2; ++mf) {
            Frag16 af;
            size_t grow =
                (size_t)(b * SEQ + srow0 + mf * 16 + l15) * EMB + h * 64 + ks * 32;
            af.q[0] = *(const uint4*)(qk + grow + kb);
            af.q[1] = *(const uint4*)(qk + grow + 16 + kb);
#pragma unroll
            for (int nf = 0; nf < 4; ++nf)
                acc[mf][nf] = __builtin_amdgcn_wmma_f32_16x16x32_bf16(
                    false, af.v, false, bfr[nf].v, (short)0, acc[mf][nf], false, false);
            accden[mf] = __builtin_amdgcn_wmma_f32_16x16x32_bf16(
                false, af.v, false, bden.v, (short)0, accden[mf], false, false);
        }
    }

#pragma unroll
    for (int mf = 0; mf < 2; ++mf) {
#pragma unroll
        for (int v = 0; v < 8; ++v) {
            float dv = __shfl(accden[mf][v], (lane & 16), 32);
            float inv = 1.f / (dv + 1e-6f);
            int row = srow0 + mf * 16 + v + ((lane >> 4) << 3);
#pragma unroll
            for (int nf = 0; nf < 4; ++nf) {
                int e = nf * 16 + l15;
                y[(size_t)(b * SEQ + row) * EMB + h * 64 + e] =
                    f32_to_bf16(acc[mf][nf][v] * inv);
            }
        }
    }
}

// ---------------------------------------------------------------------------
extern "C" void kernel_launch(void* const* d_in, const int* in_sizes, int n_in,
                              void* d_out, int out_size, void* d_ws, size_t ws_size,
                              hipStream_t stream) {
    (void)in_sizes; (void)n_in; (void)out_size; (void)ws_size;
    const float* x  = (const float*)d_in[0];
    const float* Wq = (const float*)d_in[1];
    const float* bq = (const float*)d_in[2];
    const float* Wk = (const float*)d_in[3];
    const float* bk = (const float*)d_in[4];
    const float* Wv = (const float*)d_in[5];
    const float* bv = (const float*)d_in[6];
    const float* Wo = (const float*)d_in[7];
    const float* bo = (const float*)d_in[8];
    float* out = (float*)d_out;

    char* ws = (char*)d_ws;
    size_t off = 0;
    auto alloc = [&](size_t bytes) -> void* {
        void* p = ws + off;
        off += (bytes + 255) & ~(size_t)255;
        return p;
    };
    const size_t nX = (size_t)MTOT * EMB;   // 33,554,432
    const size_t nW = (size_t)EMB * EMB;    // 1,048,576

    unsigned short* xb  = (unsigned short*)alloc(nX * 2);
    unsigned short* WqB = (unsigned short*)alloc(nW * 2);
    unsigned short* WkB = (unsigned short*)alloc(nW * 2);
    unsigned short* WvB = (unsigned short*)alloc(nW * 2);
    unsigned short* WoB = (unsigned short*)alloc(nW * 2);
    unsigned short* qb  = (unsigned short*)alloc(nX * 2);
    unsigned short* kb  = (unsigned short*)alloc(nX * 2);
    unsigned short* vb  = (unsigned short*)alloc(nX * 2);   // reused as y buffer
    unsigned short* kvb = (unsigned short*)alloc((size_t)64 * 64 * 64 * 2);
    unsigned short* ksb = (unsigned short*)alloc((size_t)64 * 64 * 2);
    unsigned short* yb  = vb;  // v dead after kv_reduce; reuse for y

    // 1) conversions to bf16
    cvt_f32_bf16_kernel<<<16384, 256, 0, stream>>>(x,  xb,  (int)nX);
    cvt_f32_bf16_kernel<<<1024,  256, 0, stream>>>(Wq, WqB, (int)nW);
    cvt_f32_bf16_kernel<<<1024,  256, 0, stream>>>(Wk, WkB, (int)nW);
    cvt_f32_bf16_kernel<<<1024,  256, 0, stream>>>(Wv, WvB, (int)nW);
    cvt_f32_bf16_kernel<<<1024,  256, 0, stream>>>(Wo, WoB, (int)nW);

    // 2) QKV projections (bf16 out)
    dim3 ggrid(EMB / 128, MTOT / 128);  // (8, 256)
    gemm_bias_kernel<1><<<ggrid, 256, 0, stream>>>(xb, WqB, bq, qb, MTOT, EMB, EMB);
    gemm_bias_kernel<1><<<ggrid, 256, 0, stream>>>(xb, WkB, bk, kb, MTOT, EMB, EMB);
    gemm_bias_kernel<1><<<ggrid, 256, 0, stream>>>(xb, WvB, bv, vb, MTOT, EMB, EMB);

    // 3) exp feature map on q, k (in place). segs = MTOT*HEADS = 524288
    feature_exp_kernel<<<65536, 256, 0, stream>>>(qb);
    feature_exp_kernel<<<65536, 256, 0, stream>>>(kb);

    // 4) per-head kv + ksum reduction
    kv_reduce_kernel<<<BATCH * HEADS, 256, 0, stream>>>(kb, vb, kvb, ksb);

    // 5) attention output y (overwrites v buffer)
    attn_out_kernel<<<BATCH * HEADS * 32, 256, 0, stream>>>(qb, kvb, ksb, yb);

    // 6) output projection (fp32 out -> d_out)
    gemm_bias_kernel<0><<<ggrid, 256, 0, stream>>>(yb, WoB, bo, out, MTOT, EMB, EMB);
}
